// CLFMv2_NoTemporalEmb_39436389712134
// MI455X (gfx1250) — matrix operand order repaired and checked by reading, not compiled, not run on hardware
//
#include <hip/hip_runtime.h>
#include <hip/hip_bf16.h>
#include <math.h>

typedef __attribute__((ext_vector_type(16))) __bf16 v16bf;
typedef __attribute__((ext_vector_type(8)))  __bf16 v8bf;
typedef __attribute__((ext_vector_type(8)))  float  v8f;

#define B_ 64
#define L_ 12
#define N_ 1024
#define D_ 64
#define H_ 128
#define O_ 12
#define R_ (B_*N_)     // 65536 total rows (B*N)
#define DT_ 0.25f      // 1/STEPS

__device__ __forceinline__ float sigmoidf_(float x){ return 1.f/(1.f+expf(-x)); }

template<int ACT> __device__ __forceinline__ float actf_(float v){
  if (ACT==1) return fmaxf(v, 0.f);
  if (ACT==2) return tanhf(v);
  if (ACT==3) return sigmoidf_(v);
  return v;
}

// Generic WMMA bf16 GEMM: C = act(epi(A@B + bias)).
//  NFRAG: output cols per wave / 16 (wave tile = 16 x NFRAG*16; block = 4 waves = 64 rows)
//  KT:    K tile staged in LDS per barrier pair (32 or 64; K % KT == 0)
//  ACT:   0 none, 1 relu, 2 tanh, 3 sigmoid
//  EPI:   0 none, 1 v += add[ci], 2 fe-update: v = fld + dt*(alpha*(af-fld) + (1-alpha)*v)
//  WF32 / WB16: write f32 / bf16 outputs
template<int NFRAG, int KT, int ACT, int EPI, bool WF32, bool WB16>
__global__ __launch_bounds__(128) void gemm_wmma(
    const __bf16* __restrict__ A, const __bf16* __restrict__ Bm,
    const float* __restrict__ bias, const float* __restrict__ add,
    const float* __restrict__ aux, const float* __restrict__ mixp,
    float* __restrict__ Cf, __bf16* __restrict__ Cb,
    int NACT, int K, long bsA, long bsB, long bsC)
{
  constexpr int NT  = NFRAG * 16;
  constexpr int NCH = NT / 8;                      // 8-wide chunks along n
  __shared__ __align__(32) __bf16 ldsBT[NT * KT];  // transposed B tile: [n][k], k contiguous

  const int lane = threadIdx.x & 31;
  const int wave = threadIdx.x >> 5;
  const int ln   = lane & 15;
  const int lh   = lane >> 4;

  const __bf16* Ab = A  + (long)blockIdx.z * bsA;
  const __bf16* Bb = Bm + (long)blockIdx.z * bsB;
  const long cOff  = (long)blockIdx.z * bsC;

  const int rowTile = blockIdx.y * 64 + wave * 16;
  const int colBase = blockIdx.x * NT;
  const long arowK  = (long)(rowTile + ln) * K;  // A-frag row for this lane
  const int kbase   = lh * 16;                   // A layout: lane-half k offset
  const int klh     = lh * 16;                   // B layout: lane-half k offset
  const bool vecOK  = (NACT & 7) == 0;           // 16B-aligned rows for vector staging

  v8f c[NFRAG];
  #pragma unroll
  for (int f = 0; f < NFRAG; ++f)
    #pragma unroll
    for (int i = 0; i < 8; ++i) c[f][i] = 0.f;

  for (int k0 = 0; k0 < K; k0 += KT) {
    __syncthreads();
    // Stage B tile (KT x NT) transposed into LDS; coalesced 16B global reads along n.
    for (int idx = threadIdx.x; idx < KT * NCH; idx += 128) {
      int ch = idx % NCH;
      int kk = idx / NCH;
      int n8 = ch * 8;
      int gn = colBase + n8;
      const __bf16* src = Bb + (long)(k0 + kk) * NACT + gn;
      if (vecOK && gn + 8 <= NACT) {
        v8bf v = *(const v8bf*)src;
        #pragma unroll
        for (int e = 0; e < 8; ++e) ldsBT[(n8 + e) * KT + kk] = v[e];
      } else {
        #pragma unroll
        for (int e = 0; e < 8; ++e)
          ldsBT[(n8 + e) * KT + kk] = (gn + e < NACT) ? src[e] : (__bf16)0.0f;
      }
    }
    // Prefetch next K-tile of the A stream (global_prefetch_b8); speculative, OOB-safe.
    if (k0 + KT < K)
      __builtin_prefetch(Ab + arowK + k0 + KT + kbase, 0, 0);
    __syncthreads();

    #pragma unroll
    for (int ks = 0; ks < KT; ks += 32) {
      // A fragment: 16x32 bf16, two contiguous 16B chunks per lane.
      const __bf16* ap = Ab + arowK + k0 + ks + kbase;
      v8bf alo = *(const v8bf*)ap;
      v8bf ahi = *(const v8bf*)(ap + 8);
      v16bf a;
      #pragma unroll
      for (int i = 0; i < 8; ++i) { a[i] = alo[i]; a[8 + i] = ahi[i]; }

      #pragma unroll
      for (int f = 0; f < NFRAG; ++f) {
        const v16bf b = *(const v16bf*)&ldsBT[(f * 16 + ln) * KT + ks + klh];
        c[f] = __builtin_amdgcn_wmma_f32_16x16x32_bf16(false, a, false, b,
                                                       (short)0, c[f], false, false);
      }
    }
  }

  float alpha = 0.f;
  if (EPI == 2) alpha = sigmoidf_(mixp[0]);

  #pragma unroll
  for (int f = 0; f < NFRAG; ++f) {
    int col = colBase + f * 16 + ln;
    if (col >= NACT) continue;
    #pragma unroll
    for (int r = 0; r < 8; ++r) {
      int row = rowTile + r + lh * 8;
      long ci = cOff + (long)row * NACT + col;
      float v = c[f][r];
      if (bias) v += bias[col];
      if (EPI == 1) v += add[ci];
      if (EPI == 2) {
        float fld = add[ci];    // field
        float af  = aux[ci];    // A @ field
        v = fld + (alpha * (af - fld) + (1.f - alpha) * v) * DT_;
      }
      v = actf_<ACT>(v);
      if (WF32) Cf[ci] = v;
      if (WB16) Cb[ci] = (__bf16)v;
    }
  }
}

// Fused GRU gate kernel: z = sig(fe@wz + st@uz + bz), c = tanh(fe@wh + st@uh + bh),
// state = (1-z)*state + z*c. M = R_, N = K = 64. 32 WMMAs per wave.
__global__ __launch_bounds__(128) void gru_wmma(
    const __bf16* __restrict__ feB, const __bf16* stB,
    const __bf16* __restrict__ wz, const __bf16* __restrict__ uz,
    const __bf16* __restrict__ wh, const __bf16* __restrict__ uh,
    const float* __restrict__ bz, const float* __restrict__ bh,
    float* stF, __bf16* stBout)
{
  __shared__ __align__(32) __bf16 lds[4 * 64 * 64];   // 4 weights, transposed [n][k]
  const __bf16* wsrc[4] = { wz, uz, wh, uh };
  const int lane = threadIdx.x & 31;
  const int wave = threadIdx.x >> 5;
  const int ln   = lane & 15;
  const int lh   = lane >> 4;

  // Vectorized transposed staging: 16B reads along n, scalar LDS scatter.
  for (int idx = threadIdx.x; idx < 4 * 64 * 8; idx += 128) {
    int w   = idx >> 9;           // weight matrix
    int rem = idx & 511;
    int kk  = rem >> 3;
    int n8  = (rem & 7) * 8;
    v8bf v = *(const v8bf*)(wsrc[w] + kk * 64 + n8);
    #pragma unroll
    for (int e = 0; e < 8; ++e) lds[w * 4096 + (n8 + e) * 64 + kk] = v[e];
  }
  __syncthreads();

  const int rowTile = blockIdx.y * 64 + wave * 16;
  const long arowK  = (long)(rowTile + ln) * 64;
  const int kbase   = lh * 16;
  const int klh     = lh * 16;

  v8f cz[4], ch[4];
  #pragma unroll
  for (int f = 0; f < 4; ++f)
    #pragma unroll
    for (int i = 0; i < 8; ++i) { cz[f][i] = 0.f; ch[f][i] = 0.f; }

  #pragma unroll
  for (int k0 = 0; k0 < 64; k0 += 32) {
    const __bf16* ap1 = feB + arowK + k0 + kbase;
    const __bf16* ap2 = stB + arowK + k0 + kbase;
    v8bf flo = *(const v8bf*)ap1; v8bf fhi = *(const v8bf*)(ap1 + 8);
    v8bf slo = *(const v8bf*)ap2; v8bf shi = *(const v8bf*)(ap2 + 8);
    v16bf afe, ast;
    #pragma unroll
    for (int i = 0; i < 8; ++i) {
      afe[i] = flo[i]; afe[8 + i] = fhi[i];
      ast[i] = slo[i]; ast[8 + i] = shi[i];
    }
    #pragma unroll
    for (int f = 0; f < 4; ++f) {
      int nb = (f * 16 + ln) * 64 + k0 + klh;
      v16bf bwz = *(const v16bf*)&lds[0 * 4096 + nb];
      v16bf buz = *(const v16bf*)&lds[1 * 4096 + nb];
      v16bf bwh = *(const v16bf*)&lds[2 * 4096 + nb];
      v16bf buh = *(const v16bf*)&lds[3 * 4096 + nb];
      cz[f] = __builtin_amdgcn_wmma_f32_16x16x32_bf16(false, afe, false, bwz, (short)0, cz[f], false, false);
      cz[f] = __builtin_amdgcn_wmma_f32_16x16x32_bf16(false, ast, false, buz, (short)0, cz[f], false, false);
      ch[f] = __builtin_amdgcn_wmma_f32_16x16x32_bf16(false, afe, false, bwh, (short)0, ch[f], false, false);
      ch[f] = __builtin_amdgcn_wmma_f32_16x16x32_bf16(false, ast, false, buh, (short)0, ch[f], false, false);
    }
  }

  #pragma unroll
  for (int f = 0; f < 4; ++f) {
    int col = f * 16 + ln;
    #pragma unroll
    for (int r = 0; r < 8; ++r) {
      int row = rowTile + r + lh * 8;
      long ci = (long)row * 64 + col;
      float z  = sigmoidf_(cz[f][r] + bz[col]);
      float cd = tanhf(ch[f][r] + bh[col]);
      float so = stF[ci];
      float sn = (1.f - z) * so + z * cd;
      stF[ci] = sn;
      stBout[ci] = (__bf16)sn;
    }
  }
}

// Row softmax of adjacency (f32 math), emits bf16 for WMMA consumption.
__global__ __launch_bounds__(256) void softmax_bf16(const float* __restrict__ adj,
                                                    __bf16* __restrict__ out)
{
  __shared__ float red[256];
  int row = blockIdx.x;
  const float* r = adj + (long)row * N_;
  int t = threadIdx.x;
  float mx = -3.4e38f;
  for (int i = t; i < N_; i += 256) mx = fmaxf(mx, r[i]);
  red[t] = mx; __syncthreads();
  for (int s = 128; s > 0; s >>= 1) { if (t < s) red[t] = fmaxf(red[t], red[t + s]); __syncthreads(); }
  float m = red[0]; __syncthreads();
  float sm = 0.f;
  for (int i = t; i < N_; i += 256) sm += expf(r[i] - m);
  red[t] = sm; __syncthreads();
  for (int s = 128; s > 0; s >>= 1) { if (t < s) red[t] += red[t + s]; __syncthreads(); }
  float inv = 1.f / red[0];
  for (int i = t; i < N_; i += 256) out[(long)row * N_ + i] = (__bf16)(expf(r[i] - m) * inv);
}

__global__ void cvt_bf16(const float* __restrict__ s, __bf16* __restrict__ d, int n) {
  int i = blockIdx.x * 256 + threadIdx.x;
  if (i < n) d[i] = (__bf16)s[i];
}

// enc_w1 (12x128) -> zero-padded bf16 (32x128) so K%32==0.
__global__ void pad_encw1(const float* __restrict__ s, __bf16* __restrict__ d) {
  int i = blockIdx.x * 256 + threadIdx.x;
  if (i < 32 * H_) {
    int row = i / H_, col = i % H_;
    d[i] = (__bf16)((row < L_) ? s[row * H_ + col] : 0.f);
  }
}

// Pack history (B,L,N,1) -> xpad (B*N, 32) bf16 with zero pad beyond L.
__global__ void enc_prep(const float* __restrict__ hist, __bf16* __restrict__ xp) {
  int t = blockIdx.x * 256 + threadIdx.x;
  if (t >= R_) return;
  int b = t >> 10, n = t & 1023;
  #pragma unroll
  for (int l = 0; l < L_; ++l)
    xp[(long)t * 32 + l] = (__bf16)hist[((long)(b * L_ + l) << 10) + n];
  #pragma unroll
  for (int l = L_; l < 32; ++l) xp[(long)t * 32 + l] = (__bf16)0.0f;
}

// (B*N, O) row-major -> (B, O, N, 1)
__global__ void out_tr(const float* __restrict__ tmp, float* __restrict__ out) {
  int t = blockIdx.x * 256 + threadIdx.x;
  if (t >= B_ * O_ * N_) return;
  int n = t & 1023;
  int o = (t >> 10) % O_;
  int b = t / (O_ * N_);
  out[t] = tmp[((long)(b * N_ + n)) * O_ + o];
}

extern "C" void kernel_launch(void* const* d_in, const int* in_sizes, int n_in,
                              void* d_out, int out_size, void* d_ws, size_t ws_size,
                              hipStream_t stream)
{
  (void)in_sizes; (void)n_in; (void)out_size; (void)ws_size;
  const float* hist   = (const float*)d_in[0];
  const float* enc_w1 = (const float*)d_in[5];
  const float* enc_b1 = (const float*)d_in[6];
  const float* enc_w2 = (const float*)d_in[7];
  const float* enc_b2 = (const float*)d_in[8];
  const float* adj    = (const float*)d_in[9];
  const float* pde_w1 = (const float*)d_in[10];
  const float* pde_b1 = (const float*)d_in[11];
  const float* pde_w2 = (const float*)d_in[12];
  const float* pde_b2 = (const float*)d_in[13];
  const float* ss_wz  = (const float*)d_in[14];
  const float* ss_uz  = (const float*)d_in[15];
  const float* ss_bz  = (const float*)d_in[16];
  const float* ss_wh  = (const float*)d_in[17];
  const float* ss_uh  = (const float*)d_in[18];
  const float* ss_bh  = (const float*)d_in[19];
  const float* ss_wo  = (const float*)d_in[20];
  const float* ss_bo  = (const float*)d_in[21];
  const float* dec_w1 = (const float*)d_in[22];
  const float* dec_b1 = (const float*)d_in[23];
  const float* dec_w2 = (const float*)d_in[24];
  const float* dec_b2 = (const float*)d_in[25];
  const float* pmix   = (const float*)d_in[26];

  char* ws = (char*)d_ws;
  size_t off = 0;
  auto alloc = [&](size_t bytes) -> char* {
    char* p = ws + off; off += (bytes + 255) & ~size_t(255); return p;
  };
  __bf16* A_b    = (__bf16*)alloc((size_t)N_ * N_ * 2);
  __bf16* xpad   = (__bf16*)alloc((size_t)R_ * 32 * 2);
  float*  fieldF = (float*) alloc((size_t)R_ * D_ * 4);
  __bf16* fieldB = (__bf16*)alloc((size_t)R_ * D_ * 2);
  float*  stateF = (float*) alloc((size_t)R_ * D_ * 4);
  __bf16* stateB = (__bf16*)alloc((size_t)R_ * D_ * 2);
  float*  AF     = (float*) alloc((size_t)R_ * D_ * 4);
  float*  feF    = (float*) alloc((size_t)R_ * D_ * 4);
  __bf16* feB    = (__bf16*)alloc((size_t)R_ * D_ * 2);
  __bf16* hidB   = (__bf16*)alloc((size_t)R_ * H_ * 2);
  float*  otmp   = (float*) alloc((size_t)R_ * O_ * 4);
  __bf16* w_e1   = (__bf16*)alloc(32 * H_ * 2);
  __bf16* w_e2   = (__bf16*)alloc(H_ * D_ * 2);
  __bf16* w_p1   = (__bf16*)alloc(D_ * H_ * 2);
  __bf16* w_p2   = (__bf16*)alloc(H_ * D_ * 2);
  __bf16* w_z    = (__bf16*)alloc(D_ * D_ * 2);
  __bf16* u_z    = (__bf16*)alloc(D_ * D_ * 2);
  __bf16* w_h    = (__bf16*)alloc(D_ * D_ * 2);
  __bf16* u_h    = (__bf16*)alloc(D_ * D_ * 2);
  __bf16* w_o    = (__bf16*)alloc(D_ * D_ * 2);
  __bf16* w_d1   = (__bf16*)alloc(D_ * H_ * 2);
  __bf16* w_d2   = (__bf16*)alloc(H_ * O_ * 2);

  auto cvt = [&](const float* s, __bf16* d, int n) {
    cvt_bf16<<<(n + 255) / 256, 256, 0, stream>>>(s, d, n);
  };
  pad_encw1<<<(32 * H_ + 255) / 256, 256, 0, stream>>>(enc_w1, w_e1);
  cvt(enc_w2, w_e2, H_ * D_);
  cvt(pde_w1, w_p1, D_ * H_);
  cvt(pde_w2, w_p2, H_ * D_);
  cvt(ss_wz, w_z, D_ * D_);  cvt(ss_uz, u_z, D_ * D_);
  cvt(ss_wh, w_h, D_ * D_);  cvt(ss_uh, u_h, D_ * D_);
  cvt(ss_wo, w_o, D_ * D_);
  cvt(dec_w1, w_d1, D_ * H_);
  cvt(dec_w2, w_d2, H_ * O_);

  softmax_bf16<<<N_, 256, 0, stream>>>(adj, A_b);
  hipMemsetAsync(stateF, 0, (size_t)R_ * D_ * 4, stream);
  hipMemsetAsync(stateB, 0, (size_t)R_ * D_ * 2, stream);
  enc_prep<<<R_ / 256, 256, 0, stream>>>(hist, xpad);

  dim3 blk(128, 1, 1);

  // Encoder: hid = relu(xpad @ enc_w1p + b1); field = hid @ enc_w2 + b2
  gemm_wmma<4, 32, 1, 0, false, true><<<dim3(2, R_ / 64, 1), blk, 0, stream>>>(
      xpad, w_e1, enc_b1, nullptr, nullptr, nullptr, nullptr, hidB, H_, 32, 0, 0, 0);
  gemm_wmma<4, 64, 0, 0, true, true><<<dim3(1, R_ / 64, 1), blk, 0, stream>>>(
      hidB, w_e2, enc_b2, nullptr, nullptr, nullptr, fieldF, fieldB, D_, H_, 0, 0, 0);

  for (int s = 0; s < 4; ++s) {
    // AF = A @ field  (batched, adjacency shared via batch-stride 0)
    gemm_wmma<4, 64, 0, 0, true, false><<<dim3(1, N_ / 64, B_), blk, 0, stream>>>(
        A_b, fieldB, nullptr, nullptr, nullptr, nullptr, AF, nullptr, D_, N_,
        0, (long)N_ * D_, (long)N_ * D_);
    // hid = tanh(field @ pde_w1 + b1)
    gemm_wmma<4, 64, 2, 0, false, true><<<dim3(2, R_ / 64, 1), blk, 0, stream>>>(
        fieldB, w_p1, pde_b1, nullptr, nullptr, nullptr, nullptr, hidB, H_, D_, 0, 0, 0);
    // fe = field + dt*(alpha*(AF - field) + (1-alpha)*(hid @ pde_w2 + b2))
    gemm_wmma<4, 64, 0, 2, true, true><<<dim3(1, R_ / 64, 1), blk, 0, stream>>>(
        hidB, w_p2, pde_b2, fieldF, AF, pmix, feF, feB, D_, H_, 0, 0, 0);
    // GRU state update (fused 4-GEMM + gates)
    gru_wmma<<<dim3(1, R_ / 64, 1), blk, 0, stream>>>(
        feB, stateB, w_z, u_z, w_h, u_h, ss_bz, ss_bh, stateF, stateB);
    // field = fe + state @ ss_wo + bo
    gemm_wmma<4, 64, 0, 1, true, true><<<dim3(1, R_ / 64, 1), blk, 0, stream>>>(
        stateB, w_o, ss_bo, feF, nullptr, nullptr, fieldF, fieldB, D_, D_, 0, 0, 0);
  }

  // Decoder: hid = relu(field @ dec_w1 + b1); otmp = hid @ dec_w2 + b2 (N=12 guarded)
  gemm_wmma<4, 64, 1, 0, false, true><<<dim3(2, R_ / 64, 1), blk, 0, stream>>>(
      fieldB, w_d1, dec_b1, nullptr, nullptr, nullptr, nullptr, hidB, H_, D_, 0, 0, 0);
  gemm_wmma<1, 64, 0, 0, true, false><<<dim3(1, R_ / 64, 1), blk, 0, stream>>>(
      hidB, w_d2, dec_b2, nullptr, nullptr, nullptr, otmp, nullptr, O_, H_, 0, 0, 0);

  out_tr<<<(B_ * O_ * N_ + 255) / 256, 256, 0, stream>>>(otmp, (float*)d_out);
}
